// ExternalMemory_2645699855026
// MI455X (gfx1250) — matrix operand order repaired and checked by reading, not compiled, not run on hardware
//
#include <hip/hip_runtime.h>
#include <hip/hip_bf16.h>

typedef __attribute__((ext_vector_type(16))) __bf16        v16bf_t;
typedef __attribute__((ext_vector_type(8)))  float         v8f;
typedef __attribute__((ext_vector_type(4))) unsigned int   v4u;
typedef __attribute__((ext_vector_type(8)))  int           i32x8;
typedef __attribute__((ext_vector_type(4)))  int           i32x4;

union V16 { v16bf_t v; v4u q[2]; unsigned short s[16]; };

#if defined(__has_builtin)
#if __has_builtin(__builtin_amdgcn_tensor_load_to_lds) && \
    __has_builtin(__builtin_amdgcn_s_wait_tensorcnt)
#define USE_TDM 1
#endif
#endif
#ifndef USE_TDM
#define USE_TDM 0
#endif

__device__ __forceinline__ unsigned short f2bf(float f) {
  unsigned int u = __float_as_uint(f);
  u += 0x7FFFu + ((u >> 16) & 1u);           // round-to-nearest-even
  return (unsigned short)(u >> 16);
}

__device__ __forceinline__ v8f wmma_bf16(v16bf_t a, v16bf_t b, v8f c) {
  // (neg_a, A, neg_b, B, c_mod, C, reuse_a, reuse_b)
  return __builtin_amdgcn_wmma_f32_16x16x32_bf16(false, a, false, b, (short)0, c,
                                                 false, false);
}

constexpr int   Bq     = 2048;
constexpr int   Mslots = 65536;
constexpr int   D      = 128;
constexpr int   SPL    = 16;               // M splits (flash-decoding style)
constexpr int   QGRP   = 16;               // 2048 / 128 queries per workgroup
constexpr int   CHUNK  = Mslots / SPL;     // 4096 keys per split
constexpr int   NIT    = CHUNK / 32;       // 128 key-block iterations
constexpr float LR     = 0.1f;

#if USE_TDM
// ---------------------------------------------------------------------------
// TDM: async DMA of a 32x128 bf16 tile (rows of keysn) into LDS with hardware
// padding: insert 4 dwords after every 64 dwords -> LDS row stride 136 shorts,
// matching the bank-conflict-free layout the compute side expects.
// D# layout per CDNA5 ISA ch.8 (group0: control/addr, group1: dims/strides).
// 6-arg builtin form: (g0, g1, g2, g3, g4, cpol).
// ---------------------------------------------------------------------------
__device__ __forceinline__ void tdm_load_keys_block(const unsigned short* gsrc,
                                                    unsigned ldsByteOff) {
  unsigned long long ga = (unsigned long long)gsrc;
  v4u g0;
  g0[0] = 1u;                                            // count=1, user desc
  g0[1] = ldsByteOff;                                    // lds_addr
  g0[2] = (unsigned)(ga & 0xFFFFFFFFull);                // global_addr[31:0]
  g0[3] = (unsigned)((ga >> 32) & 0x01FFFFFFull)         // global_addr[56:32]
        | 0x80000000u;                                   // type=2 (image)
  i32x8 g1;
  g1[0] = (int)((1u << 16)        // data_size = 1 -> 2 bytes
              | (1u << 20)        // pad_enable
              | (5u << 22)        // pad_interval: 64 dwords
              | (3u << 25));      // pad_amount:   4 dwords
  g1[1] = (int)(128u << 16);      // tensor_dim0 = 128 (elements)
  g1[2] = 0;                      // tensor_dim0 hi / tensor_dim1 lo16 (65536->0)
  g1[3] = (int)((128u << 16) | 1u); // tensor_dim1 hi16 = 1; tile_dim0 = 128
  g1[4] = 32;                     // tile_dim1 = 32; tile_dim2 = 0
  g1[5] = 128;                    // tensor_dim0_stride = 128
  g1[6] = 0;                      // stride hi / tensor_dim1_stride lo (unused, 2D)
  g1[7] = 0;
  i32x4 z4 = {0, 0, 0, 0};        // groups 2/3: unused for 2D tensors
  i32x8 z8 = {0, 0, 0, 0, 0, 0, 0, 0};
  __builtin_amdgcn_tensor_load_to_lds(g0, g1, z4, z4, z8, 0);
}
#endif

// ---------------------------------------------------------------------------
// Prep: row-normalize (cosine eps semantics) + convert to bf16. 1 wave / row.
// ---------------------------------------------------------------------------
__global__ __launch_bounds__(256) void k_norm_bf16(const float* __restrict__ src,
                                                   unsigned short* __restrict__ dst,
                                                   int nrows, int doNorm) {
  const int wv = threadIdx.x >> 5, lane = threadIdx.x & 31;
  const int r = blockIdx.x * 8 + wv;
  if (r >= nrows) return;
  const float* row = src + (size_t)r * D;
  float x0 = row[lane], x1 = row[lane + 32], x2 = row[lane + 64], x3 = row[lane + 96];
  float rn = 1.0f;
  if (doNorm) {
    float ss = x0 * x0 + x1 * x1 + x2 * x2 + x3 * x3;
#pragma unroll
    for (int off = 16; off; off >>= 1) ss += __shfl_xor(ss, off, 32);
    rn = 1.0f / fmaxf(sqrtf(ss), 1e-8f);
  }
  unsigned short* d = dst + (size_t)r * D;
  d[lane]      = f2bf(x0 * rn);
  d[lane + 32] = f2bf(x1 * rn);
  d[lane + 64] = f2bf(x2 * rn);
  d[lane + 96] = f2bf(x3 * rn);
}

// ---------------------------------------------------------------------------
// Fused flash-attention read + write-key argmax over one M-split.
// Block: 256 thr (8 waves). Wave = 16-query tile. 24 WMMAs / 32 keys / wave.
// Keys blocks DMAd by the TDM into a double-buffered padded LDS tile, one
// block ahead of compute (TENSORcnt-pipelined).
// ---------------------------------------------------------------------------
__global__ __launch_bounds__(256) void k_attn(
    const unsigned short* __restrict__ qn,     // normalized query  bf16 [B][128]
    const unsigned short* __restrict__ kn,     // normalized key    bf16 [B][128]
    const unsigned short* __restrict__ keysn,  // normalized keys   bf16 [M][128]
    const unsigned short* __restrict__ valsbf, // values            bf16 [M][128]
    float* __restrict__ pacc, float* __restrict__ pm, float* __restrict__ pl,
    float* __restrict__ pbv, int* __restrict__ pbi) {
  __shared__ __align__(16) unsigned short sk[2][32][136];  // keys, double-buffered
  __shared__ __align__(16) unsigned short svT[128][40];    // values, transposed
  __shared__ __align__(16) unsigned short pbuf[8][16][40]; // per-wave P staging

  const int tid  = threadIdx.x;
  const int wv   = tid >> 5;
  const int lane = tid & 31;
  const int l15  = lane & 15;
  const int half = lane >> 4;
  const int sel8 = half * 8, sel16 = half * 16;
  const int qg   = blockIdx.x;
  const int spl  = blockIdx.y;
  const int kb0  = spl * CHUNK;

  // Preload A operands for Q-sim and K-sim (ISA 16-bit A layout: lane = M row,
  // K chunks {sel8..sel8+7, 16+sel8..16+sel8+7} per 32-dim step).
  const int qrow = qg * 128 + wv * 16 + l15;
  V16 aq[4], akk[4];
#pragma unroll
  for (int t = 0; t < 4; ++t) {
    const unsigned short* p  = qn + (size_t)qrow * D + t * 32 + sel8;
    aq[t].q[0]  = *(const v4u*)(p);
    aq[t].q[1]  = *(const v4u*)(p + 16);
    const unsigned short* p2 = kn + (size_t)qrow * D + t * 32 + sel8;
    akk[t].q[0] = *(const v4u*)(p2);
    akk[t].q[1] = *(const v4u*)(p2 + 16);
  }

  const v8f vzero = {};
  v8f acc[8];
#pragma unroll
  for (int c = 0; c < 8; ++c) acc[c] = vzero;
  float mstat[8], lstat[8], bestv[8];
  int besti[8];
#pragma unroll
  for (int j = 0; j < 8; ++j) {
    mstat[j] = -3.0e38f; lstat[j] = 0.f; bestv[j] = -3.0e38f; besti[j] = 0;
  }

#if USE_TDM
  const unsigned skoff0 = (unsigned)(size_t)&sk[0][0][0];
  const unsigned skoff1 = (unsigned)(size_t)&sk[1][0][0];
  if (wv == 0)  // kick off block 0 before the loop
    tdm_load_keys_block(keysn + (size_t)kb0 * D, skoff0);
#endif

  for (int it = 0; it < NIT; ++it) {
    const int kb  = kb0 + it * 32;
    const int cur = it & 1;
    __syncthreads();  // all waves done reading previous buffers

#if USE_TDM
    // DMA next keys block into the other buffer while we compute this one.
    if (wv == 0 && it + 1 < NIT)
      tdm_load_keys_block(keysn + (size_t)(kb + 32) * D,
                          cur ? skoff0 : skoff1);
#else
    // Fallback: cooperative vector-load of the keys block.
#pragma unroll
    for (int rep = 0; rep < 2; ++rep) {
      int li = tid + rep * 256;
      int row = li >> 4, c4 = li & 15;
      *(v4u*)&sk[cur][row][c4 * 8] =
          *(const v4u*)(keysn + (size_t)(kb + row) * D + c4 * 8);
    }
#endif

    // ---- cooperative stage: values block transposed (dim-major) ----
    {
      int k = tid & 31, db = (tid >> 5) * 16;
      const unsigned short* vr = valsbf + (size_t)(kb + k) * D + db;
      V16 tmp;
      tmp.q[0] = *(const v4u*)vr;
      tmp.q[1] = *(const v4u*)(vr + 8);
#pragma unroll
      for (int e = 0; e < 16; ++e) svT[db + e][k] = tmp.s[e];
    }
    if (kb + 64 <= Mslots)  // prefetch next values block (global_prefetch_b8)
      __builtin_prefetch(valsbf + (size_t)(kb + 32 + (tid >> 3)) * D + (tid & 7) * 16, 0, 0);

#if USE_TDM
    if (wv == 0) {
      // TDM completes in order per wave: <=1 outstanding means current block
      // has landed; at the final iteration drain fully.
      if (it + 1 < NIT) __builtin_amdgcn_s_wait_tensorcnt(1);
      else              __builtin_amdgcn_s_wait_tensorcnt(0);
    }
#endif
    __syncthreads();  // publish sk[cur] + svT to all waves

    // ---- sim tiles: 2 column chunks x 4 K-steps; B shared by Q-sim & K-sim --
    v8f Sq[2], Skk[2];
    Sq[0] = vzero; Sq[1] = vzero; Skk[0] = vzero; Skk[1] = vzero;
#pragma unroll
    for (int nc = 0; nc < 2; ++nc) {
#pragma unroll
      for (int t = 0; t < 4; ++t) {
        V16 bt;
        const unsigned short* src = &sk[cur][nc * 16 + l15][t * 32 + sel8];
        bt.q[0] = *(const v4u*)(src);
        bt.q[1] = *(const v4u*)(src + 16);
        Sq[nc]  = wmma_bf16(aq[t].v,  bt.v, Sq[nc]);
        Skk[nc] = wmma_bf16(akk[t].v, bt.v, Skk[nc]);
      }
    }

    // ---- write-key argmax (strict > keeps first max, matching jnp.argmax) --
#pragma unroll
    for (int nc = 0; nc < 2; ++nc)
#pragma unroll
      for (int j = 0; j < 8; ++j) {
        float v = Skk[nc][j];
        if (v > bestv[j]) { bestv[j] = v; besti[j] = kb + nc * 16 + l15; }
      }

    // ---- online softmax: per-row (C layout: VGPR j = row sel8+j, col = l15) -
    float nm[8], rs[8];
#pragma unroll
    for (int j = 0; j < 8; ++j) nm[j] = fmaxf(Sq[0][j], Sq[1][j]);
#pragma unroll
    for (int off = 1; off < 16; off <<= 1)
#pragma unroll
      for (int j = 0; j < 8; ++j) nm[j] = fmaxf(nm[j], __shfl_xor(nm[j], off, 32));
#pragma unroll
    for (int j = 0; j < 8; ++j) {
      float newm = fmaxf(mstat[j], nm[j]);
      float corr = __expf(mstat[j] - newm);
      mstat[j]   = newm;
      lstat[j]  *= corr;
#pragma unroll
      for (int c = 0; c < 8; ++c) acc[c][j] *= corr;
      float p0 = __expf(Sq[0][j] - newm);
      float p1 = __expf(Sq[1][j] - newm);
      rs[j] = p0 + p1;
      pbuf[wv][sel8 + j][l15]      = f2bf(p0);
      pbuf[wv][sel8 + j][16 + l15] = f2bf(p1);
    }
#pragma unroll
    for (int off = 1; off < 16; off <<= 1)
#pragma unroll
      for (int j = 0; j < 8; ++j) rs[j] += __shfl_xor(rs[j], off, 32);
#pragma unroll
    for (int j = 0; j < 8; ++j) lstat[j] += rs[j];

    // ---- read P back in A-operand layout (DS ops in-order within wave) -----
    V16 pA;
    const unsigned short* pr = &pbuf[wv][l15][sel8];
    pA.q[0] = *(const v4u*)(pr);
    pA.q[1] = *(const v4u*)(pr + 16);

    // ---- acc(16x128) += P(16x32) x V(32x128): 8 WMMAs ----
#pragma unroll
    for (int c = 0; c < 8; ++c) {
      V16 bv;
      const unsigned short* vr2 = &svT[c * 16 + l15][sel8];
      bv.q[0] = *(const v4u*)(vr2);
      bv.q[1] = *(const v4u*)(vr2 + 16);
      acc[c] = wmma_bf16(pA.v, bv.v, acc[c]);
    }
  }

  // ---- epilogue: store split partials ----
  const int base = spl * Bq;
  const int rwg  = qg * 128 + wv * 16 + sel8;  // first of this half's 8 rows
#pragma unroll
  for (int c = 0; c < 8; ++c)
#pragma unroll
    for (int j = 0; j < 8; ++j)
      pacc[(size_t)(base + rwg + j) * D + c * 16 + l15] = acc[c][j];

#pragma unroll
  for (int j = 0; j < 8; ++j) {
    float v = bestv[j]; int i = besti[j];
#pragma unroll
    for (int off = 1; off < 16; off <<= 1) {
      float ov = __shfl_xor(v, off, 32);
      int   oi = __shfl_xor(i, off, 32);
      if (ov > v || (ov == v && oi < i)) { v = ov; i = oi; }
    }
    if (l15 == 0) {
      pm [base + rwg + j] = mstat[j];
      pl [base + rwg + j] = lstat[j];
      pbv[base + rwg + j] = v;
      pbi[base + rwg + j] = i;
    }
  }
}

// ---------------------------------------------------------------------------
// Combine M-splits: softmax merge + global argmax. 1 wave / query row.
// ---------------------------------------------------------------------------
__global__ __launch_bounds__(256) void k_combine(
    const float* __restrict__ pacc, const float* __restrict__ pm,
    const float* __restrict__ pl, const float* __restrict__ pbv,
    const int* __restrict__ pbi, float* __restrict__ outR,
    int* __restrict__ idxOut) {
  const int wv = threadIdx.x >> 5, lane = threadIdx.x & 31;
  const int r = blockIdx.x * 8 + wv;
  float ms[SPL], w[SPL];
  float gm = -3.0e38f;
#pragma unroll
  for (int s = 0; s < SPL; ++s) { ms[s] = pm[s * Bq + r]; gm = fmaxf(gm, ms[s]); }
  float L = 0.f;
#pragma unroll
  for (int s = 0; s < SPL; ++s) { w[s] = __expf(ms[s] - gm); L += pl[s * Bq + r] * w[s]; }
  const float invL = 1.0f / L;
#pragma unroll
  for (int kq = 0; kq < 4; ++kq) {
    const int d = lane + kq * 32;
    float o = 0.f;
    for (int s = 0; s < SPL; ++s)
      o += pacc[((size_t)s * Bq + r) * D + d] * w[s];
    outR[(size_t)r * D + d] = o * invL;
  }
  if (lane == 0) {
    float bv = -3.0e38f; int bi = 0x7fffffff;
    for (int s = 0; s < SPL; ++s) {
      float v = pbv[s * Bq + r]; int i = pbi[s * Bq + r];
      if (v > bv || (v == bv && i < bi)) { bv = v; bi = i; }
    }
    idxOut[r] = bi;
  }
}

// ---------------------------------------------------------------------------
// new_keys/new_values: bulk copy + deterministic last-write-wins scatter.
// ---------------------------------------------------------------------------
__global__ void k_copy(const v4u* __restrict__ ki, const v4u* __restrict__ vi,
                       v4u* __restrict__ ko, v4u* __restrict__ vo, int n) {
  int i = blockIdx.x * blockDim.x + threadIdx.x;
  if (i < n) { ko[i] = ki[i]; vo[i] = vi[i]; }
}
__global__ void k_winit(int* w) { w[blockIdx.x * 256 + threadIdx.x] = -1; }
__global__ void k_scatter(const int* __restrict__ idx, int* __restrict__ w) {
  int b = blockIdx.x * 256 + threadIdx.x;
  atomicMax(&w[idx[b]], b);                  // largest b wins = last write wins
}
__global__ void k_update(const int* __restrict__ idx, const int* __restrict__ w,
                         const float* __restrict__ keys,
                         const float* __restrict__ values,
                         const float* __restrict__ key,
                         const float* __restrict__ value,
                         float* __restrict__ outK, float* __restrict__ outV) {
  const int b = blockIdx.x;
  const int m = idx[b];
  if (w[m] != b) return;
  const int d = threadIdx.x;
  const float ok = keys[(size_t)m * D + d], ov = values[(size_t)m * D + d];
  outK[(size_t)m * D + d] = ok + LR * (key[(size_t)b * D + d] - ok);
  outV[(size_t)m * D + d] = ov + LR * (value[(size_t)b * D + d] - ov);
}

// ---------------------------------------------------------------------------
extern "C" void kernel_launch(void* const* d_in, const int* in_sizes, int n_in,
                              void* d_out, int out_size, void* d_ws, size_t ws_size,
                              hipStream_t stream) {
  (void)in_sizes; (void)n_in; (void)out_size; (void)ws_size;
  const float* query  = (const float*)d_in[0];
  const float* key    = (const float*)d_in[1];
  const float* value  = (const float*)d_in[2];
  const float* keys   = (const float*)d_in[3];
  const float* values = (const float*)d_in[4];

  float* outR = (float*)d_out;                 // [2048,128]
  float* outK = outR + (size_t)Bq * D;         // [65536,128]
  float* outV = outK + (size_t)Mslots * D;     // [65536,128]

  char* ws = (char*)d_ws;
  size_t off = 0;
  auto alloc = [&](size_t bytes) {
    void* p = ws + off;
    off = (off + bytes + 255) & ~(size_t)255;
    return p;
  };
  unsigned short* qn     = (unsigned short*)alloc((size_t)Bq * D * 2);
  unsigned short* kn     = (unsigned short*)alloc((size_t)Bq * D * 2);
  unsigned short* keysn  = (unsigned short*)alloc((size_t)Mslots * D * 2);
  unsigned short* valsbf = (unsigned short*)alloc((size_t)Mslots * D * 2);
  float* pacc = (float*)alloc((size_t)SPL * Bq * D * 4);
  float* pm   = (float*)alloc((size_t)SPL * Bq * 4);
  float* pl   = (float*)alloc((size_t)SPL * Bq * 4);
  float* pbv  = (float*)alloc((size_t)SPL * Bq * 4);
  int*   pbi  = (int*)alloc((size_t)SPL * Bq * 4);
  int*   idx  = (int*)alloc((size_t)Bq * 4);
  int*   win  = (int*)alloc((size_t)Mslots * 4);

  k_norm_bf16<<<Bq / 8,     256, 0, stream>>>(query,  qn,     Bq,     1);
  k_norm_bf16<<<Bq / 8,     256, 0, stream>>>(key,    kn,     Bq,     1);
  k_norm_bf16<<<Mslots / 8, 256, 0, stream>>>(keys,   keysn,  Mslots, 1);
  k_norm_bf16<<<Mslots / 8, 256, 0, stream>>>(values, valsbf, Mslots, 0);

  dim3 g(QGRP, SPL);  // 256 workgroups
  k_attn<<<g, 256, 0, stream>>>(qn, kn, keysn, valsbf, pacc, pm, pl, pbv, pbi);
  k_combine<<<Bq / 8, 256, 0, stream>>>(pacc, pm, pl, pbv, pbi, outR, idx);

  const int n4 = Mslots * D / 4;
  k_copy<<<n4 / 256, 256, 0, stream>>>((const v4u*)keys, (const v4u*)values,
                                       (v4u*)outK, (v4u*)outV, n4);
  k_winit<<<Mslots / 256, 256, 0, stream>>>(win);
  k_scatter<<<Bq / 256, 256, 0, stream>>>(idx, win);
  k_update<<<Bq, D, 0, stream>>>(idx, win, keys, values, key, value, outK, outV);
}
